// RazerLinear_10170482557277
// MI455X (gfx1250) — matrix operand order repaired and checked by reading, not compiled, not run on hardware
//
#include <hip/hip_runtime.h>
#include <hip/hip_fp16.h>

// ---------------------------------------------------------------------------
// Types for WMMA fragments (gfx1250, wave32)
// ---------------------------------------------------------------------------
typedef _Float16 v16h __attribute__((ext_vector_type(16)));
typedef _Float16 v8h  __attribute__((ext_vector_type(8)));
typedef float    v8f  __attribute__((ext_vector_type(8)));

struct __attribute__((aligned(8))) h4 { _Float16 a, b, c, d; };

// Four RaZeR candidate level sets, pre-sorted ascending (matches np.sort rows).
__constant__ float FP4_LEVELS[4][16] = {
    {-6.f,-4.f,-3.f,-2.f,-1.5f,-1.f,-0.5f,0.f,0.5f,1.f,1.5f,2.f,3.f,4.f,5.f,6.f},
    {-6.f,-4.f,-3.f,-2.f,-1.5f,-1.f,-0.5f,0.f,0.5f,1.f,1.5f,2.f,3.f,4.f,6.f,8.f},
    {-6.f,-5.f,-4.f,-3.f,-2.f,-1.5f,-1.f,-0.5f,0.f,0.5f,1.f,1.5f,2.f,3.f,4.f,6.f},
    {-8.f,-6.f,-4.f,-3.f,-2.f,-1.5f,-1.f,-0.5f,0.f,0.5f,1.f,1.5f,2.f,3.f,4.f,6.f},
};

// Nearest level, ties -> lower (left) level, matching the reference's
// where(|v-right| < |v-left|, right, left).
__device__ __forceinline__ float nearest_level(const float* lv, float v) {
    float best = lv[0];
    float bd   = fabsf(v - lv[0]);
#pragma unroll
    for (int i = 1; i < 16; ++i) {
        float d = fabsf(v - lv[i]);
        if (d < bd) { bd = d; best = lv[i]; }
    }
    return best;
}

// ---------------------------------------------------------------------------
// Kernel 1: fake-quantize W (groups of 128 along K), emit f16 W_dq.
// One wave32 per group; each lane owns 4 contiguous elements.
// ---------------------------------------------------------------------------
__global__ void razer_quant_w(const float* __restrict__ W,
                              _Float16* __restrict__ Wh,
                              int numGroups) {
    const int g    = (blockIdx.x * blockDim.x + threadIdx.x) >> 5;
    const int lane = threadIdx.x & 31;
    if (g >= numGroups) return;

    const float4 xv = *(const float4*)(W + (size_t)g * 128 + lane * 4);
    float x[4] = {xv.x, xv.y, xv.z, xv.w};

    float mn = fminf(fminf(x[0], x[1]), fminf(x[2], x[3]));
    float mx = fmaxf(fmaxf(x[0], x[1]), fmaxf(x[2], x[3]));
#pragma unroll
    for (int off = 16; off; off >>= 1) {
        mn = fminf(mn, __shfl_xor(mn, off, 32));
        mx = fmaxf(mx, __shfl_xor(mx, off, 32));
    }

    float bestErr = __builtin_inff();
    int   bestOpt = 0;
    float bestScale = 1.f;
#pragma unroll
    for (int opt = 0; opt < 4; ++opt) {
        const float* lv = FP4_LEVELS[opt];
        const float fmin = lv[0], fmax = lv[15];
        const float scale = fmaxf(mx / fmax, mn / fmin);
        float err = 0.f;
#pragma unroll
        for (int j = 0; j < 4; ++j) {
            float xs = fminf(fmaxf(x[j] / scale, fmin), fmax);
            float q  = nearest_level(lv, xs);
            float d  = x[j] - q * scale;
            err += d * d;
        }
#pragma unroll
        for (int off = 16; off; off >>= 1) err += __shfl_xor(err, off, 32);
        if (err < bestErr) { bestErr = err; bestOpt = opt; bestScale = scale; }
    }

    const float* lv = FP4_LEVELS[bestOpt];
    const float fmin = lv[0], fmax = lv[15];
    h4 o;
    _Float16* op = &o.a;
#pragma unroll
    for (int j = 0; j < 4; ++j) {
        float xs = fminf(fmaxf(x[j] / bestScale, fmin), fmax);
        float q  = nearest_level(lv, xs);
        op[j]    = (_Float16)(q * bestScale);
    }
    *(h4*)(Wh + (size_t)g * 128 + lane * 4) = o;
}

// ---------------------------------------------------------------------------
// Kernel 2: x fp32 -> f16 (halves GEMM read traffic, feeds the f16 WMMA pipe)
// ---------------------------------------------------------------------------
__global__ void cvt_f32_to_f16(const float* __restrict__ X,
                               _Float16* __restrict__ Xh,
                               size_t n4) {
    size_t i = blockIdx.x * (size_t)blockDim.x + threadIdx.x;
    if (i >= n4) return;
    float4 v = ((const float4*)X)[i];
    h4 o = {(_Float16)v.x, (_Float16)v.y, (_Float16)v.z, (_Float16)v.w};
    ((h4*)Xh)[i] = o;
}

// ---------------------------------------------------------------------------
// Kernel 3: C[M,N] = A[M,K] * Bw[N,K]^T + bias, f16 inputs, f32 accumulate.
// Block tile 128x128, BK=32, 8 waves (wave32), double-buffered LDS fed by
// CDNA5 async copies (global_load_async_to_lds_b128, ASYNCcnt-tracked).
// Each wave: 4x2 grid of 16x16 tiles -> 8x v_wmma_f32_16x16x32_f16 per K-step.
// ---------------------------------------------------------------------------
#define BM 128
#define BN 128
#define BK 32
#define LDT 40          // LDS row pitch in halfs: 80B = 20 dwords, conflict-free b128
#define TILE_HALFS (BM * LDT)           // 5120 halfs = 10240 B per tile buffer
#define BUF_BYTES  (TILE_HALFS * 2)     // 10240 B

__global__ __launch_bounds__(256)
void razer_gemm_wmma(const _Float16* __restrict__ A,   // [M,K]
                     const _Float16* __restrict__ Bw,  // [N,K]
                     const float* __restrict__ bias,   // [N]
                     float* __restrict__ C,            // [M,N]
                     int M, int N, int K) {
    __shared__ _Float16 lA[2][TILE_HALFS];
    __shared__ _Float16 lB[2][TILE_HALFS];

    const int tid  = threadIdx.x;
    const int lane = tid & 31;
    const int wave = tid >> 5;
    const int wm   = wave >> 2;   // 0..1  (M direction, 64 rows each)
    const int wn   = wave & 3;    // 0..3  (N direction, 32 cols each)
    const int m0   = blockIdx.y * BM;
    const int n0   = blockIdx.x * BN;

    const int lm  = lane & 15;
    const int hg  = lane >> 4;     // half-group: 0 for lanes 0-15, 1 for 16-31
    const int ksA = hg * 8;        // A-matrix K base (ISA 16-bit A layout)
    const int ksB = hg * 8;        // B-matrix K base (same layout as A)

    const int NK = K / BK;

    // ---- async global->LDS staging: each thread moves 2 x b128 per tile ----
    // chunk c: row = c>>2 (0..127), 16B-column q = c&3
    const int r0 = tid >> 2,          q0 = tid & 3;
    const int r1 = (tid + 256) >> 2,  q1 = (tid + 256) & 3;

    // Lane-varying global byte offsets (loop-invariant; K-advance goes in SGPR base)
    const unsigned gaOff0 = (unsigned)(((size_t)(m0 + r0) * K + q0 * 8) * 2);
    const unsigned gaOff1 = (unsigned)(((size_t)(m0 + r1) * K + q1 * 8) * 2);
    const unsigned gbOff0 = (unsigned)(((size_t)(n0 + r0) * K + q0 * 8) * 2);
    const unsigned gbOff1 = (unsigned)(((size_t)(n0 + r1) * K + q1 * 8) * 2);

    // LDS byte addresses for both double-buffer stages (low 32 bits of the
    // generic pointer are the wave-relative LDS address, ISA 10.2).
    unsigned ldsA0[2], ldsA1[2], ldsB0[2], ldsB1[2];
#pragma unroll
    for (int b = 0; b < 2; ++b) {
        ldsA0[b] = (unsigned)(size_t)&lA[b][r0 * LDT + q0 * 8];
        ldsA1[b] = (unsigned)(size_t)&lA[b][r1 * LDT + q1 * 8];
        ldsB0[b] = (unsigned)(size_t)&lB[b][r0 * LDT + q0 * 8];
        ldsB1[b] = (unsigned)(size_t)&lB[b][r1 * LDT + q1 * 8];
    }

    auto async_tile = [&](int kt, int b) {
        const _Float16* sA = A  + (size_t)kt * BK;  // uniform K advance (SGPR)
        const _Float16* sB = Bw + (size_t)kt * BK;
        asm volatile(
            "global_load_async_to_lds_b128 %0, %2, %4\n\t"
            "global_load_async_to_lds_b128 %1, %3, %4"
            :: "v"(ldsA0[b]), "v"(ldsA1[b]), "v"(gaOff0), "v"(gaOff1), "s"(sA)
            : "memory");
        asm volatile(
            "global_load_async_to_lds_b128 %0, %2, %4\n\t"
            "global_load_async_to_lds_b128 %1, %3, %4"
            :: "v"(ldsB0[b]), "v"(ldsB1[b]), "v"(gbOff0), "v"(gbOff1), "s"(sB)
            : "memory");
    };

    v8f acc[4][2];
#pragma unroll
    for (int t = 0; t < 4; ++t)
#pragma unroll
        for (int u = 0; u < 2; ++u) acc[t][u] = (v8f)0.f;

    async_tile(0, 0);
    asm volatile("s_wait_asynccnt 0x0" ::: "memory");
    __syncthreads();

    int buf = 0;
    for (int kt = 0; kt < NK; ++kt) {
        // Prefetch next K-tile into the other buffer while we compute.
        if (kt + 1 < NK) async_tile(kt + 1, buf ^ 1);

        union Frag { v16h v; v8h h[2]; };
        Frag fa[4], fb[2];
#pragma unroll
        for (int t = 0; t < 4; ++t) {
            const _Float16* p = &lA[buf][(wm * 64 + t * 16 + lm) * LDT + ksA];
            fa[t].h[0] = *(const v8h*)p;          // K = ksA .. ksA+7
            fa[t].h[1] = *(const v8h*)(p + 16);   // K = ksA+16 .. ksA+23
        }
#pragma unroll
        for (int u = 0; u < 2; ++u) {
            const _Float16* p = &lB[buf][(wn * 32 + u * 16 + lm) * LDT + ksB];
            fb[u].h[0] = *(const v8h*)p;          // K = ksB .. ksB+7
            fb[u].h[1] = *(const v8h*)(p + 16);   // K = ksB+16 .. ksB+23
        }
#pragma unroll
        for (int t = 0; t < 4; ++t)
#pragma unroll
            for (int u = 0; u < 2; ++u)
                acc[t][u] = __builtin_amdgcn_wmma_f32_16x16x32_f16(
                    false, fa[t].v, false, fb[u].v,
                    (short)0, acc[t][u], false, false);

        // Drain this wave's async copies (compute above hid the latency),
        // then make all waves' LDS writes visible.
        asm volatile("s_wait_asynccnt 0x0\n\ts_wait_xcnt 0x0" ::: "memory");
        __syncthreads();
        buf ^= 1;
    }

    // Epilogue: C/D layout -> M = r + 8*hg, N = lane&15 per 16x16 tile.
#pragma unroll
    for (int t = 0; t < 4; ++t) {
#pragma unroll
        for (int u = 0; u < 2; ++u) {
            const int col   = n0 + wn * 32 + u * 16 + lm;
            const int rbase = m0 + wm * 64 + t * 16 + hg * 8;
            const float bv  = bias[col];
#pragma unroll
            for (int r = 0; r < 8; ++r)
                C[(size_t)(rbase + r) * N + col] = acc[t][u][r] + bv;
        }
    }
}

// ---------------------------------------------------------------------------
// Launch
// ---------------------------------------------------------------------------
extern "C" void kernel_launch(void* const* d_in, const int* in_sizes, int n_in,
                              void* d_out, int out_size, void* d_ws, size_t ws_size,
                              hipStream_t stream) {
    const float* x = (const float*)d_in[0];
    const float* W = (const float*)d_in[1];
    const float* b = (const float*)d_in[2];
    float* out = (float*)d_out;

    const int D_OUT = in_sizes[2];                       // 4096
    const int D_IN  = in_sizes[1] / D_OUT;               // 4096
    const long long M = (long long)in_sizes[0] / D_IN;   // 8192
    const int N = D_OUT, K = D_IN;

    _Float16* Wh = (_Float16*)d_ws;                      // [N,K] f16
    _Float16* Xh = Wh + (size_t)N * K;                   // [M,K] f16

    const int numGroups = (int)(((size_t)N * K) / 128);
    razer_quant_w<<<(numGroups + 7) / 8, 256, 0, stream>>>(W, Wh, numGroups);

    const size_t n4 = ((size_t)M * K) / 4;
    cvt_f32_to_f16<<<(unsigned)((n4 + 255) / 256), 256, 0, stream>>>(x, Xh, n4);

    dim3 grid((unsigned)(N / BN), (unsigned)(M / BM));
    razer_gemm_wmma<<<grid, 256, 0, stream>>>(Xh, Wh, b, out, (int)M, N, K);
}
